// CCCrossLayerAttentionB_76227079569758
// MI455X (gfx1250) — matrix operand order repaired and checked by reading, not compiled, not run on hardware
//
#include <hip/hip_runtime.h>

// ---------------------------------------------------------------------------
// CDNA5 (gfx1250) CCCrossLayerAttention.
// Roofline: ~137 GFLOP, ~1 GB HBM traffic -> memory-bound at bf16 WMMA rates
// (v_wmma_f32_16x16x32_bf16, f32 accumulation). One generic strided batched
// GEMM kernel covers every einsum; all M/N are multiples of 32, all K of 32
// -> no tails. Each wave computes a 32x32 tile (2x2 WMMA tiles, 4 wmma per
// K-step sharing 2 A + 2 B fragments). Templated fast paths use b128 vector
// loads when the K dimension is contiguous. All addressing is 32-bit (largest
// buffer is 37.7M elements), avoiding wave32 64-bit VALU mults in hot loops.
// ---------------------------------------------------------------------------

typedef __attribute__((ext_vector_type(16))) __bf16 v16bf;
typedef __attribute__((ext_vector_type(8)))  float  v8f;
typedef __attribute__((ext_vector_type(4)))  float  f32x4;

#define MODE_STORE    0   // Y = acc (+bias)
#define MODE_ACCUM    1   // Y += acc
#define MODE_CC_FINAL 2   // Y = gamma*(acc + T) + Y   (criss-cross residual)

#define NEG_INF_F (-1000000000.0f)

__device__ __forceinline__ __bf16 f2bf(float f) {
  // round-to-nearest-even f32 -> bf16
  unsigned u = __builtin_bit_cast(unsigned, f);
  u += 0x7FFFu + ((u >> 16) & 1u);
  unsigned short h = (unsigned short)(u >> 16);
  return __builtin_bit_cast(__bf16, h);
}

struct GemmParams {
  const float* A; const float* B; float* Y;
  const float* bias;      // per-row (channel) bias or nullptr
  const float* T;         // temp operand for MODE_CC_FINAL (same geometry as Y)
  const float* gammaPtr;  // device scalar for MODE_CC_FINAL
  int a_off0, b_off0, y_off0;            // constant base offsets (elements)
  int a_bo, a_bi, b_bo, b_bi, y_bo, y_bi;// batch outer/inner strides
  int a_ms, a_ks;                        // A strides: rows (m), K
  int b_ks, b_ns;                        // B strides: K, cols (n)
  int y_ms, y_ns;                        // Y strides: rows, cols
  int inner;                             // inner batch count (>=1)
  int M, N, K, mode;
};

// Lane roles used by this code (A/B share one contiguous K order; C/D per CDNA5 ISA):
//   A (16x32): lane holds row m = lane%16; element e holds
//              K = e + 16*(lane/16)
//   B (32x16): lane holds col n = lane%16; element e holds K = e + 16*(lane/16)
//   C/D (16x16 f32): VGPR r -> m = r + 8*(lane/16), n = lane%16

// Load one A-layout fragment with contiguous K (one aligned 16-float run).
__device__ __forceinline__ v16bf load_a_contig(const float* pa) {
  const f32x4 r0 = *(const f32x4*)(pa + 0);
  const f32x4 r1 = *(const f32x4*)(pa + 4);
  const f32x4 r2 = *(const f32x4*)(pa + 8);
  const f32x4 r3 = *(const f32x4*)(pa + 12);
  v16bf a;
#pragma unroll
  for (int e = 0; e < 4; ++e) {
    a[e]      = f2bf(r0[e]);
    a[e + 4]  = f2bf(r1[e]);
    a[e + 8]  = f2bf(r2[e]);
    a[e + 12] = f2bf(r3[e]);
  }
  return a;
}

// Load one B-layout fragment with contiguous K (one aligned 16-float run).
__device__ __forceinline__ v16bf load_b_contig(const float* pb) {
  const f32x4 q0 = *(const f32x4*)(pb + 0);
  const f32x4 q1 = *(const f32x4*)(pb + 4);
  const f32x4 q2 = *(const f32x4*)(pb + 8);
  const f32x4 q3 = *(const f32x4*)(pb + 12);
  v16bf b;
#pragma unroll
  for (int e = 0; e < 4; ++e) {
    b[e]      = f2bf(q0[e]);
    b[e + 4]  = f2bf(q1[e]);
    b[e + 8]  = f2bf(q2[e]);
    b[e + 12] = f2bf(q3[e]);
  }
  return b;
}

template <bool AC, bool BC>
__global__ void __launch_bounds__(256)
gemm_bf16_wmma(GemmParams p, int total_tiles) {
  const int wid = blockIdx.x * (blockDim.x >> 5) + (threadIdx.x >> 5);
  if (wid >= total_tiles) return;          // whole-wave exit: EXEC all-1s
  const int lane = threadIdx.x & 31;
  const int half = lane >> 4;
  const int l16  = lane & 15;

  const int tiles_m2 = p.M >> 5;           // wave covers 2 m-tiles (32 rows)
  const int tiles_n2 = p.N >> 5;           // and 2 n-tiles (32 cols)
  const int per_batch = tiles_m2 * tiles_n2;
  const int batch = wid / per_batch;
  const int rem   = wid - batch * per_batch;
  const int tm = rem % tiles_m2;
  const int tn = rem / tiles_m2;
  const int bo = batch / p.inner;
  const int bi = batch - bo * p.inner;

  const float* Ap = p.A + p.a_off0 + bo * p.a_bo + bi * p.a_bi;
  const float* Bp = p.B + p.b_off0 + bo * p.b_bo + bi * p.b_bi;
  const int yoff  = p.y_off0 + bo * p.y_bo + bi * p.y_bi;
  float* Yp = p.Y + yoff;

  const int aRow0 = (tm * 32 + l16) * p.a_ms;       // lane's A row, m-tile 0
  const int aRow1 = aRow0 + 16 * p.a_ms;            // m-tile 1 (m+16)
  const int bCol0 = (tn * 32 + l16) * p.b_ns;       // lane's B col, n-tile 0
  const int bCol1 = bCol0 + 16 * p.b_ns;            // n-tile 1 (n+16)

  // loop-invariant gather offsets for the strided (non-contiguous) paths
  int aoff[16], boff[16];
  if (!AC) {
#pragma unroll
    for (int e = 0; e < 16; ++e)
      aoff[e] = (e + half * 16) * p.a_ks;
  }
  if (!BC) {
#pragma unroll
    for (int e = 0; e < 16; ++e)
      boff[e] = (e + half * 16) * p.b_ks;
  }

  v8f acc00 = {}, acc01 = {}, acc10 = {}, acc11 = {};
  int ka = 0, kb = 0;                                // k0*a_ks / k0*b_ks
  for (int k0 = 0; k0 < p.K; k0 += 32, ka += 32 * p.a_ks, kb += 32 * p.b_ks) {
    if (k0 + 32 < p.K) {   // gfx1250 global_prefetch_b8 of next K block
      __builtin_prefetch(Bp + bCol0 + kb + (32 + 16 * half) * p.b_ks, 0, 1);
      __builtin_prefetch(Ap + aRow0 + ka + (32 + 8 * half) * p.a_ks, 0, 1);
    }

    v16bf b0, b1;
    if (BC) {              // K contiguous: 4x global_load_b128 per fragment
      b0 = load_b_contig(Bp + bCol0 + kb + 16 * half);
      b1 = load_b_contig(Bp + bCol1 + kb + 16 * half);
    } else {
#pragma unroll
      for (int e = 0; e < 16; ++e) {
        b0[e] = f2bf(Bp[bCol0 + kb + boff[e]]);
        b1[e] = f2bf(Bp[bCol1 + kb + boff[e]]);
      }
    }

    v16bf a0, a1;
    if (AC) {
      a0 = load_a_contig(Ap + aRow0 + ka + 16 * half);
      a1 = load_a_contig(Ap + aRow1 + ka + 16 * half);
    } else {
#pragma unroll
      for (int e = 0; e < 16; ++e) {
        a0[e] = f2bf(Ap[aRow0 + ka + aoff[e]]);
        a1[e] = f2bf(Ap[aRow1 + ka + aoff[e]]);
      }
    }

    acc00 = __builtin_amdgcn_wmma_f32_16x16x32_bf16(
        false, a0, false, b0, (short)0, acc00, false, false);
    acc01 = __builtin_amdgcn_wmma_f32_16x16x32_bf16(
        false, a0, false, b1, (short)0, acc01, false, false);
    acc10 = __builtin_amdgcn_wmma_f32_16x16x32_bf16(
        false, a1, false, b0, (short)0, acc10, false, false);
    acc11 = __builtin_amdgcn_wmma_f32_16x16x32_bf16(
        false, a1, false, b1, (short)0, acc11, false, false);
  }

  const float* Tp   = (p.mode == MODE_CC_FINAL) ? (p.T + yoff) : nullptr;
  const float gamma = (p.mode == MODE_CC_FINAL) ? *p.gammaPtr : 0.0f;
  const int n0 = tn * 32 + l16;
  const int n1 = n0 + 16;
#pragma unroll
  for (int r = 0; r < 8; ++r) {
    const int m0 = tm * 32 + r + half * 8;    // m-tile 0 row
    const int m1 = m0 + 16;                   // m-tile 1 row
    const int y00 = m0 * p.y_ms + n0 * p.y_ns;
    const int y01 = m0 * p.y_ms + n1 * p.y_ns;
    const int y10 = m1 * p.y_ms + n0 * p.y_ns;
    const int y11 = m1 * p.y_ms + n1 * p.y_ns;
    float v00 = acc00[r], v01 = acc01[r], v10 = acc10[r], v11 = acc11[r];
    if (p.bias) {
      const float bb0 = p.bias[m0], bb1 = p.bias[m1];
      v00 += bb0; v01 += bb0; v10 += bb1; v11 += bb1;
    }
    if (p.mode == MODE_ACCUM) {
      v00 += Yp[y00]; v01 += Yp[y01]; v10 += Yp[y10]; v11 += Yp[y11];
    } else if (p.mode == MODE_CC_FINAL) {
      v00 = gamma * (v00 + Tp[y00]) + Yp[y00];
      v01 = gamma * (v01 + Tp[y01]) + Yp[y01];
      v10 = gamma * (v10 + Tp[y10]) + Yp[y10];
      v11 = gamma * (v11 + Tp[y11]) + Yp[y11];
    }
    Yp[y00] = v00; Yp[y01] = v01; Yp[y10] = v10; Yp[y11] = v11;
  }
}

// ------------------------- bilinear 2x upsample ----------------------------
__global__ void __launch_bounds__(256)
upsample_bilinear(const float* __restrict__ src, float* __restrict__ dst,
                  int Hin, int Win, int Hout, int Wout, long long total) {
  const long long idx = (long long)blockIdx.x * blockDim.x + threadIdx.x;
  if (idx >= total) return;
  const int x = (int)(idx % Wout);
  const int y = (int)((idx / Wout) % Hout);
  const int bc = (int)(idx / ((long long)Wout * Hout));
  const float sy = (y + 0.5f) * ((float)Hin / (float)Hout) - 0.5f;
  const float sx = (x + 0.5f) * ((float)Win / (float)Wout) - 0.5f;
  const int y0 = (int)floorf(sy), x0 = (int)floorf(sx);
  const float fy = sy - (float)y0, fx = sx - (float)x0;
  const int y0c = min(max(y0, 0), Hin - 1), y1c = min(max(y0 + 1, 0), Hin - 1);
  const int x0c = min(max(x0, 0), Win - 1), x1c = min(max(x0 + 1, 0), Win - 1);
  const float* s = src + (long long)bc * Hin * Win;
  const float v00 = s[y0c * Win + x0c], v01 = s[y0c * Win + x1c];
  const float v10 = s[y1c * Win + x0c], v11 = s[y1c * Win + x1c];
  dst[idx] = (1.0f - fy) * ((1.0f - fx) * v00 + fx * v01)
           +          fy * ((1.0f - fx) * v10 + fx * v11);
}

// ---------------- softmax over concat([eH, eW]) with diag mask -------------
// att layout: [b][h][w][192]; position H'==h of the first 96 gets NEG_INF.
__global__ void __launch_bounds__(256)
softmax_cc(float* __restrict__ att, int H, int W, int total_pixels) {
  const int wave = blockIdx.x * (blockDim.x >> 5) + (threadIdx.x >> 5);
  if (wave >= total_pixels) return;
  const int lane = threadIdx.x & 31;
  const int h = (wave / W) % H;
  float* a = att + (long long)wave * 192;
  float vals[6];
  float m = -3.0e38f;
#pragma unroll
  for (int i = 0; i < 6; ++i) {
    const int j = lane + 32 * i;
    float v = a[j];
    if (j == h) v += NEG_INF_F;          // eye(H) mask on the eH part
    vals[i] = v;
    m = fmaxf(m, v);
  }
  for (int off = 16; off; off >>= 1) m = fmaxf(m, __shfl_xor(m, off, 32));
  float s = 0.0f;
#pragma unroll
  for (int i = 0; i < 6; ++i) { vals[i] = __expf(vals[i] - m); s += vals[i]; }
  for (int off = 16; off; off >>= 1) s += __shfl_xor(s, off, 32);
  const float inv = 1.0f / s;
#pragma unroll
  for (int i = 0; i < 6; ++i) a[lane + 32 * i] = vals[i] * inv;
}

// ------------------- batchnorm: deterministic 2-stage ----------------------
__global__ void __launch_bounds__(256)
bn_reduce(const float* __restrict__ x, float* __restrict__ part, int C, int HW) {
  // blockIdx.x = b*C + c ; reduce one (b,c) image plane
  const long long plane = (long long)blockIdx.x * HW;
  float s1 = 0.0f, s2 = 0.0f;
  for (int i = threadIdx.x; i < HW; i += blockDim.x) {
    const float v = x[plane + i];
    s1 += v; s2 += v * v;
  }
  __shared__ float sh1[256], sh2[256];
  sh1[threadIdx.x] = s1; sh2[threadIdx.x] = s2; __syncthreads();
  for (int off = 128; off; off >>= 1) {
    if ((int)threadIdx.x < off) {
      sh1[threadIdx.x] += sh1[threadIdx.x + off];
      sh2[threadIdx.x] += sh2[threadIdx.x + off];
    }
    __syncthreads();
  }
  if (threadIdx.x == 0) {
    part[blockIdx.x] = sh1[0];
    part[gridDim.x + blockIdx.x] = sh2[0];
  }
}

__global__ void __launch_bounds__(256)
bn_finalize(const float* __restrict__ part, float* __restrict__ coef,
            const float* __restrict__ bn_scale, const float* __restrict__ bn_bias,
            int C, int NB, float invN) {
  const int c = blockIdx.x * blockDim.x + threadIdx.x;
  if (c >= C) return;
  float s1 = 0.0f, s2 = 0.0f;
  const int BC = NB * C;
  for (int b = 0; b < NB; ++b) {
    s1 += part[b * C + c];
    s2 += part[BC + b * C + c];
  }
  const float mean = s1 * invN;
  const float var  = s2 * invN - mean * mean;
  const float sf   = rsqrtf(var + 1e-5f) * bn_scale[c];
  coef[c]     = sf;
  coef[C + c] = bn_bias[c] - mean * sf;
}

__global__ void __launch_bounds__(256)
bn_apply_relu(float* __restrict__ y, const float* __restrict__ coef,
              int C, int HW, long long total) {
  const long long idx = (long long)blockIdx.x * blockDim.x + threadIdx.x;
  if (idx >= total) return;
  const int c = (int)((idx / HW) % C);
  const float v = y[idx] * coef[c] + coef[C + c];
  y[idx] = v > 0.0f ? v : 0.0f;
}

// ---------------------------------------------------------------------------
extern "C" void kernel_launch(void* const* d_in, const int* in_sizes, int n_in,
                              void* d_out, int out_size, void* d_ws, size_t ws_size,
                              hipStream_t stream) {
  const float* low    = (const float*)d_in[0];
  const float* high   = (const float*)d_in[1];
  const float* w1     = (const float*)d_in[2];
  const float* b1     = (const float*)d_in[3];
  const float* w2     = (const float*)d_in[4];
  const float* b2     = (const float*)d_in[5];
  const float* qw     = (const float*)d_in[6];
  const float* qb     = (const float*)d_in[7];
  const float* kw     = (const float*)d_in[8];
  const float* kb     = (const float*)d_in[9];
  const float* vw     = (const float*)d_in[10];
  const float* vb     = (const float*)d_in[11];
  const float* gammaP = (const float*)d_in[12];
  const float* bw     = (const float*)d_in[13];
  const float* bnsc   = (const float*)d_in[14];
  const float* bnbi   = (const float*)d_in[15];
  float* out = (float*)d_out;

  constexpr int Bn = 8, C = 512, H = 96, W = 96, HW = 96 * 96;
  constexpr int Cin = 256, Ci = 32;
  constexpr long long SZ_HF  = (long long)Bn * C   * HW;  // 37,748,736
  constexpr long long SZ_CF  = (long long)Bn * Cin * HW;  // 18,874,368
  constexpr long long SZ_Q   = (long long)Bn * Ci  * HW;  //  2,359,296
  constexpr long long SZ_ATT = (long long)Bn * HW * 192;  // 14,155,776

  // workspace layout (f32 elements), total ~453 MB
  float* hf   = (float*)d_ws;
  float* qf   = hf  + SZ_HF;   // reused as tmpH (outH) after q is computed
  float* vf   = qf  + SZ_CF;
  float* qq   = vf  + SZ_CF;
  float* kk   = qq  + SZ_Q;
  float* vv   = kk  + SZ_Q;
  float* att  = vv  + SZ_CF;
  float* part = att + SZ_ATT;                  // Bn*C*2
  float* coef = part + (long long)Bn * C * 2;  // C*2
  float* tmpH = qf;

  auto launch_gemm = [&](const GemmParams& p, int nbatch) {
    const int tiles = (p.M / 32) * (p.N / 32) * nbatch;
    const unsigned blocks = (unsigned)((tiles + 7) / 8);  // 8 waves / block
    const bool ac = (p.a_ks == 1), bc = (p.b_ks == 1);
    if (ac && bc)       gemm_bf16_wmma<true,  true ><<<blocks, 256, 0, stream>>>(p, tiles);
    else if (ac)        gemm_bf16_wmma<true,  false><<<blocks, 256, 0, stream>>>(p, tiles);
    else if (bc)        gemm_bf16_wmma<false, true ><<<blocks, 256, 0, stream>>>(p, tiles);
    else                gemm_bf16_wmma<false, false><<<blocks, 256, 0, stream>>>(p, tiles);
  };

  // 1) hf = bilinear_resize(high, 48x48 -> 96x96)
  {
    const long long total = SZ_HF;
    upsample_bilinear<<<(unsigned)((total + 255) / 256), 256, 0, stream>>>(
        high, hf, 48, 48, 96, 96, total);
  }

  // 2) qf = conv1_w (256x512) x low + b1 ; 3) vf = conv2_w x hf + b2
  {
    GemmParams p{};
    p.A = w1; p.B = low; p.Y = qf; p.bias = b1;
    p.a_ms = C; p.a_ks = 1;
    p.b_bo = C * HW; p.b_ks = HW; p.b_ns = 1;
    p.y_bo = Cin * HW; p.y_ms = HW; p.y_ns = 1;
    p.inner = 1; p.M = Cin; p.N = HW; p.K = C; p.mode = MODE_STORE;
    launch_gemm(p, Bn);
    p.A = w2; p.bias = b2; p.B = hf; p.Y = vf;
    launch_gemm(p, Bn);
  }

  // 4) q = q_w (32x256) x qf + q_b   (constant across both criss-cross iters)
  {
    GemmParams p{};
    p.A = qw; p.B = qf; p.Y = qq; p.bias = qb;
    p.a_ms = Cin; p.a_ks = 1;
    p.b_bo = Cin * HW; p.b_ks = HW; p.b_ns = 1;
    p.y_bo = Ci * HW; p.y_ms = HW; p.y_ns = 1;
    p.inner = 1; p.M = Ci; p.N = HW; p.K = Cin; p.mode = MODE_STORE;
    launch_gemm(p, Bn);
  }

  // 5) two criss-cross iterations
  for (int it = 0; it < 2; ++it) {
    // k = k_w x vf + k_b
    {
      GemmParams p{};
      p.A = kw; p.B = vf; p.Y = kk; p.bias = kb;
      p.a_ms = Cin; p.a_ks = 1;
      p.b_bo = Cin * HW; p.b_ks = HW; p.b_ns = 1;
      p.y_bo = Ci * HW; p.y_ms = HW; p.y_ns = 1;
      p.inner = 1; p.M = Ci; p.N = HW; p.K = Cin; p.mode = MODE_STORE;
      launch_gemm(p, Bn);
    }
    // v = v_w x vf + v_b
    {
      GemmParams p{};
      p.A = vw; p.B = vf; p.Y = vv; p.bias = vb;
      p.a_ms = Cin; p.a_ks = 1;
      p.b_bo = Cin * HW; p.b_ks = HW; p.b_ns = 1;
      p.y_bo = Cin * HW; p.y_ms = HW; p.y_ns = 1;
      p.inner = 1; p.M = Cin; p.N = HW; p.K = Cin; p.mode = MODE_STORE;
      launch_gemm(p, Bn);
    }
    // eH[b,h,w,H'] = sum_c q[b,c,h,w] k[b,c,H',w]  (batched over (b,w), K=32)
    {
      GemmParams p{};
      p.A = qq; p.B = kk; p.Y = att;
      p.a_bo = Ci * HW; p.a_bi = 1; p.a_ms = W;  p.a_ks = HW;
      p.b_bo = Ci * HW; p.b_bi = 1; p.b_ks = HW; p.b_ns = W;
      p.y_bo = HW * 192; p.y_bi = 192;
      p.y_ms = W * 192;  p.y_ns = 1;
      p.inner = W; p.M = H; p.N = H; p.K = Ci; p.mode = MODE_STORE;
      launch_gemm(p, Bn * W);
    }
    // eW[b,h,w,W'] = sum_c q[b,c,h,w] k[b,c,h,W']  (batched over (b,h))
    {
      GemmParams p{};
      p.A = qq; p.B = kk; p.Y = att;
      p.a_bo = Ci * HW; p.a_bi = W; p.a_ms = 1;  p.a_ks = HW;
      p.b_bo = Ci * HW; p.b_bi = W; p.b_ks = HW; p.b_ns = 1;
      p.y_off0 = 96;
      p.y_bo = HW * 192; p.y_bi = W * 192;
      p.y_ms = 192; p.y_ns = 1;
      p.inner = H; p.M = W; p.N = W; p.K = Ci; p.mode = MODE_STORE;
      launch_gemm(p, Bn * H);
    }
    // softmax over the 192 concatenated logits (diag mask applied inline)
    {
      const int pixels = Bn * HW;
      softmax_cc<<<(unsigned)((pixels + 7) / 8), 256, 0, stream>>>(att, H, W, pixels);
    }
    // outH[b,c,h,w] = sum_H' v[b,c,H',w] aH[b,h,w,H']  -> tmpH
    {
      GemmParams p{};
      p.A = vv; p.B = att; p.Y = tmpH;
      p.a_bo = Cin * HW; p.a_bi = 1; p.a_ms = HW; p.a_ks = W;
      p.b_bo = HW * 192; p.b_bi = 192;
      p.b_ks = 1; p.b_ns = W * 192;
      p.y_bo = Cin * HW; p.y_bi = 1; p.y_ms = HW; p.y_ns = W;
      p.inner = W; p.M = Cin; p.N = H; p.K = H; p.mode = MODE_STORE;
      launch_gemm(p, Bn * W);
    }
    // outW + fused epilogue: vf = gamma*(tmpH + outW) + vf
    {
      GemmParams p{};
      p.A = vv; p.B = att; p.Y = vf; p.T = tmpH; p.gammaPtr = gammaP;
      p.a_bo = Cin * HW; p.a_bi = W; p.a_ms = HW; p.a_ks = 1;
      p.b_off0 = 96;
      p.b_bo = HW * 192; p.b_bi = W * 192;
      p.b_ks = 1; p.b_ns = 192;
      p.y_bo = Cin * HW; p.y_bi = W; p.y_ms = HW; p.y_ns = 1;
      p.inner = H; p.M = Cin; p.N = W; p.K = W; p.mode = MODE_CC_FINAL;
      launch_gemm(p, Bn * H);
    }
  }

  // 6) bottleneck conv over concat([vf, hf]) as two accumulating GEMM passes
  {
    GemmParams p{};
    p.A = bw; p.B = vf; p.Y = out;
    p.a_ms = Cin + C; p.a_ks = 1;
    p.b_bo = Cin * HW; p.b_ks = HW; p.b_ns = 1;
    p.y_bo = C * HW; p.y_ms = HW; p.y_ns = 1;
    p.inner = 1; p.M = C; p.N = HW; p.K = Cin; p.mode = MODE_STORE;
    launch_gemm(p, Bn);                       // out  = W[:, :256] x vf
    p.a_off0 = Cin; p.B = hf; p.b_bo = C * HW;
    p.K = C; p.mode = MODE_ACCUM;
    launch_gemm(p, Bn);                       // out += W[:, 256:] x hf
  }

  // 7) batchnorm (deterministic two-stage reduction) + ReLU, in place on out
  {
    bn_reduce<<<Bn * C, 256, 0, stream>>>(out, part, C, HW);
    bn_finalize<<<(C + 255) / 256, 256, 0, stream>>>(
        part, coef, bnsc, bnbi, C, Bn, 1.0f / (float)((long long)Bn * HW));
    const long long total = (long long)Bn * C * HW;
    bn_apply_relu<<<(unsigned)((total + 255) / 256), 256, 0, stream>>>(
        out, coef, C, HW, total);
  }
  (void)in_sizes; (void)n_in; (void)out_size; (void)ws_size;
}
